// MoE_56075093016699
// MI455X (gfx1250) — matrix operand order repaired and checked by reading, not compiled, not run on hardware
//
#include <hip/hip_runtime.h>
#include <math.h>

#define T_TOK   8192
#define DIM     768
#define FF      3072
#define NE      8
#define CAPACITY 2560            // int(8192*2/8*1.25)
#define NSLOT   (T_TOK * 2)      // 16384 (t,k) slots
#define MAXROWS 17408            // >= 16384 + 8*63, multiple of 64
#define MAXTILES (MAXROWS / 64)  // 272
#define LDAB    40               // LDS row stride in ushorts (80B = 5*16B, conflict-free)

typedef __attribute__((ext_vector_type(16))) __bf16 v16bf;
typedef __attribute__((ext_vector_type(8)))  __bf16 v8bf;
typedef __attribute__((ext_vector_type(8)))  float  v8f;

union FragAB { v16bf v; uint4 q[2]; unsigned short us[16]; };

// Vector fp32 -> bf16 conversion (RNE): lowers to packed v_cvt_pk_bf16_f32 with
// both operands in one instruction, no scalar half-merge chains.
__device__ __forceinline__ uint4 pack8(float4 v0, float4 v1) {
    v8f f = {v0.x, v0.y, v0.z, v0.w, v1.x, v1.y, v1.z, v1.w};
    union { v8bf b; uint4 q; } c;
    c.b = __builtin_convertvector(f, v8bf);
    return c.q;
}
__device__ __forceinline__ unsigned short f2bf(float f) {
    union { __bf16 b; unsigned short u; } c;
    c.b = (__bf16)f;
    return c.u;
}

// A fragment 16x32 bf16 (M x K), ISA 7.12.2 layout:
// lanes 0-15: M=lane, halves 0-7 -> K=0..7, halves 8-15 -> K=16..23
// lanes 16-31: M=lane-16, halves 0-7 -> K=8..15, halves 8-15 -> K=24..31
__device__ __forceinline__ v16bf load_frag_a(const unsigned short* lds, int mbase, int lane) {
    int m = mbase + (lane & 15);
    int k = (lane < 16) ? 0 : 8;
    FragAB f;
    f.q[0] = *(const uint4*)&lds[m * LDAB + k];
    f.q[1] = *(const uint4*)&lds[m * LDAB + 16 + k];
    return f.v;
}
// B fragment 32x16 bf16 (K x N): lanes 0-15 hold K=0..7,16..23,
// lanes 16-31 hold K=8..15,24..31; N = lane & 15. LDS holds B as [N][K].
__device__ __forceinline__ v16bf load_frag_b(const unsigned short* lds, int nbase, int lane) {
    int n = nbase + (lane & 15);
    int k = (lane < 16) ? 0 : 8;
    FragAB f;
    f.q[0] = *(const uint4*)&lds[n * LDAB + k];
    f.q[1] = *(const uint4*)&lds[n * LDAB + 16 + k];
    return f.v;
}

#define BF16MMA(a, b, c) \
    __builtin_amdgcn_wmma_f32_16x16x32_bf16(false, (a), false, (b), (short)0, (c), false, false)

// Stage one gathered activation row chunk (fp32 -> bf16) into LDS.
__device__ __forceinline__ void stage_x(unsigned short* As, const float* xrow,
                                        int arow, int acg, int k0) {
    uint4 st = make_uint4(0u, 0u, 0u, 0u);
    if (xrow) {
        float4 v0 = *(const float4*)(xrow + k0 + acg * 8);
        float4 v1 = *(const float4*)(xrow + k0 + acg * 8 + 4);
        st = pack8(v0, v1);
    }
    *(uint4*)&As[arow * LDAB + acg * 8] = st;
}
// Stage a 128-row x 32-col fp32 weight tile (row stride `stride`) as bf16 into LDS.
__device__ __forceinline__ void stage_w(unsigned short* Bs, const float* w,
                                        int row0, int stride, int k0, int tid) {
#pragma unroll
    for (int j = 0; j < 2; ++j) {
        int c = j * 256 + tid;
        int brow = c >> 2, bcg = c & 3;
        const float* p = w + (size_t)(row0 + brow) * stride + k0 + bcg * 8;
        *(uint4*)&Bs[brow * LDAB + bcg * 8] =
            pack8(*(const float4*)p, *(const float4*)(p + 4));
    }
}

// ---------------------------------------------------------------- init
__global__ void k_init(int* row_of_slot, int* row_token, float* row_w, int* cnt, int* cnt2) {
    int i = blockIdx.x * blockDim.x + threadIdx.x;
    if (i < NSLOT) row_of_slot[i] = -1;
    if (i < MAXROWS) { row_token[i] = -1; row_w[i] = 0.0f; }
    if (i < NE) { cnt[i] = 0; cnt2[i] = 0; }
}

// ---------------------------------------------------------------- router
// 1024 blocks x 256 threads; one wave (32 lanes) per token.
__global__ __launch_bounds__(256) void k_router(
    const float* __restrict__ x, const float* __restrict__ w_gate,
    int* __restrict__ topk_e, float* __restrict__ topk_p, float* __restrict__ partials)
{
    __shared__ float wg[NE * DIM];          // 24 KB
    __shared__ float part[8][17];
    int tid = threadIdx.x;
    for (int i = tid; i < NE * DIM; i += 256) wg[i] = w_gate[i];
    __syncthreads();

    int wv = tid >> 5, lane = tid & 31;
    int token = blockIdx.x * 8 + wv;

    float acc[NE];
#pragma unroll
    for (int e = 0; e < NE; ++e) acc[e] = 0.0f;
    const float* xr = x + (size_t)token * DIM;
    for (int d = lane; d < DIM; d += 32) {
        float xv = xr[d];
#pragma unroll
        for (int e = 0; e < NE; ++e) acc[e] += xv * wg[e * DIM + d];
    }
#pragma unroll
    for (int e = 0; e < NE; ++e)
        for (int off = 16; off; off >>= 1) acc[e] += __shfl_xor(acc[e], off, 32);

    if (lane == 0) {
        float l[NE];
#pragma unroll
        for (int e = 0; e < NE; ++e) l[e] = acc[e];
        float m = l[0]; int i0 = 0;
        for (int e = 1; e < NE; ++e) if (l[e] > m) { m = l[e]; i0 = e; }
        float m2 = -1e30f; int i1 = 0;
        for (int e = 0; e < NE; ++e) if (e != i0 && l[e] > m2) { m2 = l[e]; i1 = e; }
        float p[NE], denom = 0.0f;
#pragma unroll
        for (int e = 0; e < NE; ++e) { p[e] = __expf(l[e] - m); denom += p[e]; }
        float inv = 1.0f / denom;
#pragma unroll
        for (int e = 0; e < NE; ++e) p[e] *= inv;
        float lse = m + logf(denom);
        float p0 = p[i0], p1 = p[i1], s = p0 + p1;
        topk_e[2 * token]     = i0;
        topk_e[2 * token + 1] = i1;
        topk_p[2 * token]     = p0 / s;
        topk_p[2 * token + 1] = p1 / s;
#pragma unroll
        for (int e = 0; e < NE; ++e) part[wv][e] = p[e];
#pragma unroll
        for (int e = 0; e < NE; ++e) part[wv][8 + e] = (e == i0 || e == i1) ? 1.0f : 0.0f;
        part[wv][16] = lse * lse;
    }
    __syncthreads();
    if (tid < 17) {
        float s = 0.0f;
        for (int w = 0; w < 8; ++w) s += part[w][tid];
        partials[blockIdx.x * 17 + tid] = s;
    }
}

// ---------------------------------------------------------------- aux losses
__global__ void k_aux(const float* __restrict__ partials, float* __restrict__ out) {
    __shared__ float red[17];
    int tid = threadIdx.x;
    if (tid < 17) {
        float s = 0.0f;
        for (int b = 0; b < 1024; ++b) s += partials[b * 17 + tid];
        red[tid] = s;
    }
    __syncthreads();
    if (tid == 0) {
        float lb = 0.0f;
        for (int e = 0; e < NE; ++e) {
            float f = red[8 + e] / (float)(T_TOK * 2);
            float P = red[e] / (float)T_TOK;
            lb += f * P;
        }
        lb *= (float)NE;
        float z = red[16] / (float)T_TOK;
        out[(size_t)T_TOK * DIM] = 0.01f * lb + 0.001f * z;
    }
}

// ---------------------------------------------------------------- rank / capacity keep
// 2 blocks (one per k) x 32 threads; exact token-index-order scan via ballots.
__global__ void k_rank(const int* __restrict__ topk_e, const float* __restrict__ topk_p,
                       float* __restrict__ slot_w)
{
    int k = blockIdx.x, lane = threadIdx.x;
    int base[NE];
#pragma unroll
    for (int e = 0; e < NE; ++e) base[e] = 0;
    for (int t0 = 0; t0 < T_TOK; t0 += 32) {
        int t = t0 + lane;
        int e = topk_e[2 * t + k];
        unsigned mymask = 0;
        int cnts[NE];
#pragma unroll
        for (int ee = 0; ee < NE; ++ee) {
            unsigned m = (unsigned)__ballot(e == ee);
            if (ee == e) mymask = m;
            cnts[ee] = __popc(m);
        }
        int rank = base[e] + __popc(mymask & ((1u << lane) - 1u));
        float w = (rank < CAPACITY) ? topk_p[2 * t + k] : 0.0f;
        slot_w[2 * t + k] = w;
#pragma unroll
        for (int ee = 0; ee < NE; ++ee) base[ee] += cnts[ee];
    }
}

// ---------------------------------------------------------------- count / offsets / fill
__global__ void k_count(const int* __restrict__ topk_e, const float* __restrict__ slot_w,
                        int* __restrict__ cnt) {
    int s = blockIdx.x * blockDim.x + threadIdx.x;
    if (s < NSLOT && slot_w[s] > 0.0f) atomicAdd(&cnt[topk_e[s]], 1);
}
__global__ void k_offsets(const int* __restrict__ cnt, int* __restrict__ offs) {
    int off = 0;
    for (int e = 0; e < NE; ++e) {
        offs[e] = off;
        off += ((cnt[e] + 63) / 64) * 64;   // pad each expert to 64-row tiles
    }
    offs[NE] = off;
}
__global__ void k_fill(const int* __restrict__ topk_e, const float* __restrict__ slot_w,
                       const int* __restrict__ offs, int* __restrict__ cnt2,
                       int* __restrict__ row_of_slot, int* __restrict__ row_token,
                       float* __restrict__ row_w) {
    int s = blockIdx.x * blockDim.x + threadIdx.x;
    if (s >= NSLOT) return;
    float w = slot_w[s];
    if (w > 0.0f) {
        int e = topk_e[s];
        int pos = atomicAdd(&cnt2[e], 1);
        int row = offs[e] + pos;
        row_of_slot[s] = row;
        row_token[row] = s >> 1;
        row_w[row] = w;
    }
}

// ---------------------------------------------------------------- gate/up GEMM + SiLU -> H (bf16)
// Tile: 64 rows x 128 F-cols, BK=32, LDS double-buffered (one barrier per K-step).
// 8 waves as 2(M) x 4(N); each wave: 2x2 16x16 tiles x {gate,up} = 8 WMMAs/K-step.
__global__ __launch_bounds__(256) void k_gateup(
    const float* __restrict__ x, const float* __restrict__ wi_gate,
    const float* __restrict__ wi_up, const int* __restrict__ row_token,
    const int* __restrict__ offs, unsigned short* __restrict__ Hbuf)
{
    __shared__ int soffs[NE + 1];
    __shared__ unsigned short As[2][64 * LDAB];    //  2 x  5 KB
    __shared__ unsigned short Bg[2][128 * LDAB];   //  2 x 10 KB
    __shared__ unsigned short Bu[2][128 * LDAB];   //  2 x 10 KB
    int tid = threadIdx.x;
    if (tid <= NE) soffs[tid] = offs[tid];
    __syncthreads();
    int m0 = blockIdx.y * 64;
    if (m0 >= soffs[NE]) return;
    int e = 0;
    while (e < NE - 1 && m0 >= soffs[e + 1]) ++e;
    int f0 = blockIdx.x * 128;
    const float* wgp = wi_gate + (size_t)e * FF * DIM;
    const float* wup = wi_up   + (size_t)e * FF * DIM;

    int arow = tid >> 2, acg = tid & 3;
    int tokenA = row_token[m0 + arow];
    const float* xrow = (tokenA >= 0) ? (x + (size_t)tokenA * DIM) : (const float*)0;

    int lane = tid & 31, wv = tid >> 5;
    int wm = wv >> 2, wn = wv & 3;

    v8f accg[2][2], accu[2][2];
    v8f vz = {0.f, 0.f, 0.f, 0.f, 0.f, 0.f, 0.f, 0.f};
#pragma unroll
    for (int i = 0; i < 2; ++i)
#pragma unroll
        for (int j = 0; j < 2; ++j) { accg[i][j] = vz; accu[i][j] = vz; }

    const int NT = DIM / 32;  // 24
    stage_x(As[0], xrow, arow, acg, 0);
    stage_w(Bg[0], wgp, f0, DIM, 0, tid);
    stage_w(Bu[0], wup, f0, DIM, 0, tid);
    __syncthreads();

    for (int kt = 0; kt < NT; ++kt) {
        int cur = kt & 1;
        if (kt + 1 < NT) {
            int kn = (kt + 1) * 32;
            stage_x(As[cur ^ 1], xrow, arow, acg, kn);
            stage_w(Bg[cur ^ 1], wgp, f0, DIM, kn, tid);
            stage_w(Bu[cur ^ 1], wup, f0, DIM, kn, tid);
        }
        v16bf a0 = load_frag_a(As[cur], wm * 32, lane);
        v16bf a1 = load_frag_a(As[cur], wm * 32 + 16, lane);
        v16bf g0 = load_frag_b(Bg[cur], wn * 32, lane);
        v16bf g1 = load_frag_b(Bg[cur], wn * 32 + 16, lane);
        v16bf u0 = load_frag_b(Bu[cur], wn * 32, lane);
        v16bf u1 = load_frag_b(Bu[cur], wn * 32 + 16, lane);
        accg[0][0] = BF16MMA(a0, g0, accg[0][0]);
        accg[0][1] = BF16MMA(a0, g1, accg[0][1]);
        accg[1][0] = BF16MMA(a1, g0, accg[1][0]);
        accg[1][1] = BF16MMA(a1, g1, accg[1][1]);
        accu[0][0] = BF16MMA(a0, u0, accu[0][0]);
        accu[0][1] = BF16MMA(a0, u1, accu[0][1]);
        accu[1][0] = BF16MMA(a1, u0, accu[1][0]);
        accu[1][1] = BF16MMA(a1, u1, accu[1][1]);
        __syncthreads();
    }
    // epilogue: H = silu(gate) * up -> bf16
    int mlane = (lane & 16) ? 8 : 0;
    int nlane = lane & 15;
#pragma unroll
    for (int mi = 0; mi < 2; ++mi)
#pragma unroll
        for (int ni = 0; ni < 2; ++ni) {
            v8f g = accg[mi][ni], u = accu[mi][ni];
#pragma unroll
            for (int r = 0; r < 8; ++r) {
                float gv = g[r];
                float hv = (gv / (1.0f + __expf(-gv))) * u[r];
                int row = m0 + wm * 32 + mi * 16 + mlane + r;
                int col = f0 + wn * 32 + ni * 16 + nlane;
                Hbuf[(size_t)row * FF + col] = f2bf(hv);
            }
        }
}

// ---------------------------------------------------------------- down-proj GEMM * combine -> slotY (f32)
__global__ __launch_bounds__(256) void k_down(
    const unsigned short* __restrict__ Hbuf, const float* __restrict__ wo,
    const int* __restrict__ offs, const float* __restrict__ row_w,
    float* __restrict__ slotY)
{
    __shared__ int soffs[NE + 1];
    __shared__ unsigned short As[2][64 * LDAB];
    __shared__ unsigned short Bs[2][128 * LDAB];
    int tid = threadIdx.x;
    if (tid <= NE) soffs[tid] = offs[tid];
    __syncthreads();
    int m0 = blockIdx.y * 64;
    if (m0 >= soffs[NE]) return;
    int e = 0;
    while (e < NE - 1 && m0 >= soffs[e + 1]) ++e;
    int d0 = blockIdx.x * 128;
    const float* wop = wo + (size_t)e * DIM * FF;

    int arow = tid >> 2, acg = tid & 3;
    const unsigned short* hrow = Hbuf + (size_t)(m0 + arow) * FF;
    int lane = tid & 31, wv = tid >> 5;
    int wm = wv >> 2, wn = wv & 3;

    v8f acc[2][2];
    v8f vz = {0.f, 0.f, 0.f, 0.f, 0.f, 0.f, 0.f, 0.f};
#pragma unroll
    for (int i = 0; i < 2; ++i)
#pragma unroll
        for (int j = 0; j < 2; ++j) acc[i][j] = vz;

    const int NT = FF / 32;  // 96
    *(uint4*)&As[0][arow * LDAB + acg * 8] = *(const uint4*)&hrow[acg * 8];
    stage_w(Bs[0], wop, d0, FF, 0, tid);
    __syncthreads();

    for (int kt = 0; kt < NT; ++kt) {
        int cur = kt & 1;
        if (kt + 1 < NT) {
            int kn = (kt + 1) * 32;
            *(uint4*)&As[cur ^ 1][arow * LDAB + acg * 8] =
                *(const uint4*)&hrow[kn + acg * 8];
            stage_w(Bs[cur ^ 1], wop, d0, FF, kn, tid);
        }
        v16bf a0 = load_frag_a(As[cur], wm * 32, lane);
        v16bf a1 = load_frag_a(As[cur], wm * 32 + 16, lane);
        v16bf b0 = load_frag_b(Bs[cur], wn * 32, lane);
        v16bf b1 = load_frag_b(Bs[cur], wn * 32 + 16, lane);
        acc[0][0] = BF16MMA(a0, b0, acc[0][0]);
        acc[0][1] = BF16MMA(a0, b1, acc[0][1]);
        acc[1][0] = BF16MMA(a1, b0, acc[1][0]);
        acc[1][1] = BF16MMA(a1, b1, acc[1][1]);
        __syncthreads();
    }
    int mlane = (lane & 16) ? 8 : 0;
    int nlane = lane & 15;
#pragma unroll
    for (int mi = 0; mi < 2; ++mi)
#pragma unroll
        for (int ni = 0; ni < 2; ++ni) {
            v8f y = acc[mi][ni];
#pragma unroll
            for (int r = 0; r < 8; ++r) {
                int row = m0 + wm * 32 + mi * 16 + mlane + r;
                int col = d0 + wn * 32 + ni * 16 + nlane;
                slotY[(size_t)row * DIM + col] = y[r] * row_w[row];
            }
        }
}

// ---------------------------------------------------------------- combine slots -> output (deterministic)
__global__ void k_combine(const float* __restrict__ slotY, const int* __restrict__ row_of_slot,
                          float* __restrict__ out) {
    int t = blockIdx.x;
    int r0 = row_of_slot[2 * t], r1 = row_of_slot[2 * t + 1];
#pragma unroll
    for (int j = 0; j < 3; ++j) {
        int d = threadIdx.x + j * 256;
        float v = 0.0f;
        if (r0 >= 0) v += slotY[(size_t)r0 * DIM + d];
        if (r1 >= 0) v += slotY[(size_t)r1 * DIM + d];
        out[(size_t)t * DIM + d] = v;
    }
}

// ---------------------------------------------------------------- launch
extern "C" void kernel_launch(void* const* d_in, const int* in_sizes, int n_in,
                              void* d_out, int out_size, void* d_ws, size_t ws_size,
                              hipStream_t stream) {
    (void)in_sizes; (void)n_in; (void)out_size; (void)ws_size;
    const float* x       = (const float*)d_in[0];
    const float* w_gate  = (const float*)d_in[1];
    const float* wi_gate = (const float*)d_in[2];
    const float* wi_up   = (const float*)d_in[3];
    const float* wo      = (const float*)d_in[4];
    float* out = (float*)d_out;

    char* ws = (char*)d_ws;
    size_t off = 0;
    auto carve = [&](size_t bytes) -> void* {
        void* p = ws + off;
        off = (off + bytes + 255) & ~(size_t)255;
        return p;
    };
    int*   topk_e      = (int*)  carve((size_t)NSLOT * 4);
    float* topk_p      = (float*)carve((size_t)NSLOT * 4);
    float* slot_w      = (float*)carve((size_t)NSLOT * 4);
    int*   row_of_slot = (int*)  carve((size_t)NSLOT * 4);
    int*   row_token   = (int*)  carve((size_t)MAXROWS * 4);
    float* row_w       = (float*)carve((size_t)MAXROWS * 4);
    int*   cnt         = (int*)  carve(NE * 4);
    int*   cnt2        = (int*)  carve(NE * 4);
    int*   offs        = (int*)  carve((NE + 8) * 4);
    float* partials    = (float*)carve((size_t)1024 * 17 * 4);
    unsigned short* Hbuf = (unsigned short*)carve((size_t)MAXROWS * FF * 2);
    float* slotY       = (float*)carve((size_t)MAXROWS * DIM * 4);

    k_init<<<(MAXROWS + 255) / 256, 256, 0, stream>>>(row_of_slot, row_token, row_w, cnt, cnt2);
    k_router<<<1024, 256, 0, stream>>>(x, w_gate, topk_e, topk_p, partials);
    k_aux<<<1, 256, 0, stream>>>(partials, out);
    k_rank<<<2, 32, 0, stream>>>(topk_e, topk_p, slot_w);
    k_count<<<(NSLOT + 255) / 256, 256, 0, stream>>>(topk_e, slot_w, cnt);
    k_offsets<<<1, 1, 0, stream>>>(cnt, offs);
    k_fill<<<(NSLOT + 255) / 256, 256, 0, stream>>>(topk_e, slot_w, offs, cnt2,
                                                    row_of_slot, row_token, row_w);
    dim3 g1(FF / 128, MAXTILES);
    k_gateup<<<g1, 256, 0, stream>>>(x, wi_gate, wi_up, row_token, offs, Hbuf);
    dim3 g2(DIM / 128, MAXTILES);
    k_down<<<g2, 256, 0, stream>>>(Hbuf, wo, offs, row_w, slotY);
    k_combine<<<T_TOK, 256, 0, stream>>>(slotY, row_of_slot, out);
}
